// FaultInjectionAttentionShim_54709293417081
// MI455X (gfx1250) — compile-verified, matrix operand order checked by reading, not run on hardware
//
#include <hip/hip_runtime.h>
#include <hip/hip_bf16.h>
#include <stdint.h>

// CDNA5 / gfx1250 attention block.
// Matrix math via v_wmma_f32_16x16x32_f16 (fp32 accum); bulk straight copies
// via the Tensor Data Mover (tensor_load_to_lds + s_wait_tensorcnt).
typedef __attribute__((ext_vector_type(16))) _Float16 v16h;
typedef __attribute__((ext_vector_type(8)))  _Float16 v8h;
typedef __attribute__((ext_vector_type(8)))  float    v8f;
typedef __attribute__((ext_vector_type(4)))  unsigned int v4u;
typedef __attribute__((ext_vector_type(4)))  int      v4i;
typedef __attribute__((ext_vector_type(8)))  int      v8i;

#define S_LEN 2048
#define NHEAD 32
#define NKV   8
#define HDIM  128
#define HID   4096
#define KVHID 1024

#if __has_builtin(__builtin_amdgcn_tensor_load_to_lds) && \
    __has_builtin(__builtin_amdgcn_s_wait_tensorcnt)
#define USE_TDM 1
#else
#define USE_TDM 0
#endif

__device__ __forceinline__ v8f wmma_f16(v16h a, v16h b, v8f c) {
  // (neg_a, A, neg_b, B, c_mod, C, reuse_a, reuse_b)
  return __builtin_amdgcn_wmma_f32_16x16x32_f16(false, a, false, b, (short)0, c,
                                                false, false);
}

#if USE_TDM
// 2D f16 tile DMA: global tensor [tensor_m x tensor_k] (row stride tensor_k
// elements) -> LDS, tile [tile_m x tile_k], rows stored back-to-back.
// D# layout per CDNA5 ISA.
__device__ __forceinline__ void tdm_load_2d_f16(unsigned lds_off, const void* gptr,
                                                unsigned tensor_k, unsigned tensor_m,
                                                unsigned tile_k, unsigned tile_m) {
  unsigned long long ga = (unsigned long long)(uintptr_t)gptr;
  v4u g0;
  g0[0] = 1u;                                   // count=1 (valid), flags 0
  g0[1] = lds_off;                              // lds_addr (bytes)
  g0[2] = (unsigned)ga;                         // global_addr[31:0]
  g0[3] = (unsigned)((ga >> 32) & 0x1FFFFFFu)   // global_addr[56:32]
          | (2u << 30);                         // type=2 ("image")
  v8i g1;
  g1[0] = (int)(1u << 16);                      // data_size=1 (2 bytes)
  g1[1] = (int)((tensor_k & 0xFFFFu) << 16);    // tensor_dim0[15:0] @63:48
  g1[2] = (int)(((tensor_k >> 16) & 0xFFFFu)    // tensor_dim0[31:16]
          | ((tensor_m & 0xFFFFu) << 16));      // tensor_dim1[15:0]
  g1[3] = (int)(((tensor_m >> 16) & 0xFFFFu)    // tensor_dim1[31:16]
          | (tile_k << 16));                    // tile_dim0
  g1[4] = (int)tile_m;                          // tile_dim1 (tile_dim2=0)
  g1[5] = (int)tensor_k;                        // tensor_dim0_stride[31:0]
  g1[6] = 0;                                    // stride hi / dim1_stride lo
  g1[7] = 0;
  v4i gz = {0, 0, 0, 0};
#if defined(__clang_major__) && __clang_major__ >= 23
  v8i gz8 = {0, 0, 0, 0, 0, 0, 0, 0};
  __builtin_amdgcn_tensor_load_to_lds(g0, g1, gz, gz, gz8, 0);
#else
  __builtin_amdgcn_tensor_load_to_lds(g0, g1, gz, gz, 0);
#endif
}
#endif

// ---------------------------------------------------------------------------
// Tiled GEMM: C[M,N] = A[M,K] * B[K,N].  A fp32 or f16, B fp32, C f16 or fp32.
// Block = 256 threads = 8 waves; block tile 64(M) x 64(N); k-step 32.
// f16-A tiles staged by the Tensor Data Mover; fp32 tiles converted manually.
// ---------------------------------------------------------------------------
template <bool A_F16, bool C_F16>
__global__ __launch_bounds__(256) void gemm_tile(const void* __restrict__ Aptr,
                                                 const float* __restrict__ B,
                                                 void* __restrict__ Cptr,
                                                 int M, int N, int K) {
  __shared__ alignas(32) _Float16 sA[64][32];  // [m][k]
  __shared__ alignas(32) _Float16 sB[64][32];  // [n][k] (transposed)
  const int t = threadIdx.x;
  const int lane = t & 31, wave = t >> 5;
  const int wm = wave >> 1, wn = wave & 1;
  const int m_base = blockIdx.x * 64, n_base = blockIdx.y * 64;
  const int hl = lane >> 4;        // 0 -> K 0..15, 1 -> K 16..31 (frag split)
  const int ll = lane & 15;

  v8f acc[2] = {};
  for (int k0 = 0; k0 < K; k0 += 32) {
    // ---- stage A tile 64x32 ----
    if constexpr (A_F16) {
#if USE_TDM
      if (wave == 0) {  // one DMA for the whole tile (TDM ignores EXEC)
        const _Float16* A = (const _Float16*)Aptr;
        tdm_load_2d_f16((unsigned)(size_t)&sA[0][0],
                        &A[(size_t)m_base * K + k0],
                        (unsigned)K, (unsigned)M, 32u, 64u);
        __builtin_amdgcn_s_wait_tensorcnt(0);
      }
#else
      {
        int row = t >> 2, c8 = (t & 3) * 8;
        const _Float16* A = (const _Float16*)Aptr;
        *(v8h*)&sA[row][c8] =
            *(const v8h*)&A[(size_t)(m_base + row) * K + k0 + c8];
      }
#endif
    } else {
      int row = t >> 2, c8 = (t & 3) * 8;
      const float* A = (const float*)Aptr;
      const float4* p = (const float4*)&A[(size_t)(m_base + row) * K + k0 + c8];
      float4 a0 = p[0], a1 = p[1];
      sA[row][c8 + 0] = (_Float16)a0.x; sA[row][c8 + 1] = (_Float16)a0.y;
      sA[row][c8 + 2] = (_Float16)a0.z; sA[row][c8 + 3] = (_Float16)a0.w;
      sA[row][c8 + 4] = (_Float16)a1.x; sA[row][c8 + 5] = (_Float16)a1.y;
      sA[row][c8 + 6] = (_Float16)a1.z; sA[row][c8 + 7] = (_Float16)a1.w;
    }
    // ---- stage B tile 32x64, transposed [n][k], fp32 -> f16 ----
    {
      int kr = t >> 3, nc = (t & 7) * 8;
      const float4* p = (const float4*)&B[(size_t)(k0 + kr) * N + n_base + nc];
      float4 b0 = p[0], b1 = p[1];
      sB[nc + 0][kr] = (_Float16)b0.x; sB[nc + 1][kr] = (_Float16)b0.y;
      sB[nc + 2][kr] = (_Float16)b0.z; sB[nc + 3][kr] = (_Float16)b0.w;
      sB[nc + 4][kr] = (_Float16)b1.x; sB[nc + 5][kr] = (_Float16)b1.y;
      sB[nc + 6][kr] = (_Float16)b1.z; sB[nc + 7][kr] = (_Float16)b1.w;
      if (k0 + 32 < K)  // global_prefetch_b8 for next B panel
        __builtin_prefetch(&B[(size_t)(k0 + 32 + kr) * N + n_base + nc], 0, 1);
    }
    __syncthreads();
    v16h af = *(const v16h*)&sA[wm * 16 + ll][hl * 16];
#pragma unroll
    for (int j = 0; j < 2; ++j) {
      v16h bf = *(const v16h*)&sB[wn * 32 + j * 16 + ll][hl * 16];
      acc[j] = wmma_f16(af, bf, acc[j]);
    }
    __syncthreads();
  }
  // C/D layout: lanes 0-15 rows 0..7 (vgpr r), lanes 16-31 rows 8..15; col=ll
#pragma unroll
  for (int j = 0; j < 2; ++j) {
#pragma unroll
    for (int r = 0; r < 8; ++r) {
      int row = m_base + wm * 16 + hl * 8 + r;
      int col = n_base + wn * 32 + j * 16 + ll;
      if constexpr (C_F16)
        ((_Float16*)Cptr)[(size_t)row * N + col] = (_Float16)acc[j][r];
      else
        ((float*)Cptr)[(size_t)row * N + col] = acc[j][r];
    }
  }
}

// ---------------------------------------------------------------------------
// RoPE in-place on f16 [S, nheads, 128]; one thread per (s, h, i<64) pair.
// ---------------------------------------------------------------------------
__global__ __launch_bounds__(256) void rope_kernel(_Float16* __restrict__ buf,
                                                   const int* __restrict__ pos_ids,
                                                   int nheads, int total) {
  int idx = blockIdx.x * blockDim.x + threadIdx.x;
  if (idx >= total) return;
  int i = idx & 63;
  int h = (idx >> 6) % nheads;
  int s = idx / (nheads * 64);
  float p = (float)pos_ids[s];
  // inv_freq = 10000^(-2i/128) = exp(-i * ln(10000)/64)
  float ang = p * __expf(-(float)i * (9.21034037198f / 64.0f));
  float c = __cosf(ang), sn = __sinf(ang);
  size_t base = ((size_t)s * nheads + h) * HDIM;
  float x0 = (float)buf[base + i];
  float x1 = (float)buf[base + 64 + i];
  buf[base + i]      = (_Float16)(x0 * c - x1 * sn);
  buf[base + 64 + i] = (_Float16)(x1 * c + x0 * sn);
}

// ---------------------------------------------------------------------------
// Causal flash attention, GQA 4:1.  Grid (S/128, H); 256 threads = 8 waves,
// each wave owns 16 query rows.  Key blocks of 32: K panel staged once per
// block by the TDM (layout [key][d] == score B-fragment layout), V staged
// manually transposed, P re-laid-out through per-wave LDS.  Online softmax
// via __shfl_xor inside each 16-lane half (matches WMMA C-matrix rows).
// ---------------------------------------------------------------------------
__global__ __launch_bounds__(256) void flash_attn(const _Float16* __restrict__ q,
                                                  const _Float16* __restrict__ k,
                                                  const _Float16* __restrict__ v,
                                                  _Float16* __restrict__ attn) {
  __shared__ alignas(32) _Float16 sK[32][128];     // [key][d] K block (TDM)
  __shared__ alignas(32) _Float16 sV[128][32];     // [d][key] transposed V
  __shared__ alignas(32) _Float16 sP[8][16][32];   // per-wave P tile (A layout)
  const int h = blockIdx.y, hkv = h >> 2;
  const int t = threadIdx.x, lane = t & 31, wave = t >> 5;
  const int hl = lane >> 4, ll = lane & 15;
  const int q0w = blockIdx.x * 128 + wave * 16;
  const float scale = 0.088388347648318447f;  // 1/sqrt(128)

  // Q A-fragments for 4 K32 chunks of D (row = ll, K-halves by hl).
  v16h aq[4];
#pragma unroll
  for (int d = 0; d < 4; ++d)
    aq[d] = *(const v16h*)&q[(size_t)(q0w + ll) * HID + h * HDIM + d * 32 + hl * 16];

  v8f oacc[8] = {};
  float m[8], l[8];
#pragma unroll
  for (int r = 0; r < 8; ++r) { m[r] = -1e30f; l[r] = 0.0f; }

  const int nkb = 4 * (blockIdx.x + 1);  // cover keys up to block q_max
  for (int kb = 0; kb < nkb; ++kb) {
    const int k0 = kb * 32;
    __syncthreads();  // protect sK / sV / sP from previous iteration readers
#if USE_TDM
    if (wave == 0) {  // stage K block 32x128 with one tensor DMA
      tdm_load_2d_f16((unsigned)(size_t)&sK[0][0],
                      &k[(size_t)k0 * KVHID + hkv * HDIM],
                      (unsigned)KVHID, (unsigned)S_LEN, (unsigned)HDIM, 32u);
      __builtin_amdgcn_s_wait_tensorcnt(0);
    }
#else
    {  // cooperative K stage: straight copy [key][d]
      int key = t >> 3, d0 = (t & 7) * 16;
      *(v16h*)&sK[key][d0] =
          *(const v16h*)&k[(size_t)(k0 + key) * KVHID + hkv * HDIM + d0];
    }
#endif
    {  // cooperative V stage: 32 keys x 128 d -> sV[d][key] (transpose)
      int key = t >> 3, d0 = (t & 7) * 16;
      v16h vv = *(const v16h*)&v[(size_t)(k0 + key) * KVHID + hkv * HDIM + d0];
#pragma unroll
      for (int i = 0; i < 16; ++i) sV[d0 + i][key] = vv[i];
      if (kb + 1 < nkb) {
        __builtin_prefetch(&v[(size_t)(k0 + 32 + key) * KVHID + hkv * HDIM + d0], 0, 1);
        __builtin_prefetch(&k[(size_t)(k0 + 32 + key) * KVHID + hkv * HDIM + d0], 0, 1);
      }
    }
    __syncthreads();
    const bool active = (k0 <= q0w + 15);  // wave-uniform causal skip
    if (active) {
      // ---- scores: two 16x16 key tiles, 4 WMMAs each over D ----
      v8f sc0 = {}, sc1 = {};
#pragma unroll
      for (int d = 0; d < 4; ++d) {
        v16h bk = *(const v16h*)&sK[ll][d * 32 + hl * 16];
        sc0 = wmma_f16(aq[d], bk, sc0);
      }
#pragma unroll
      for (int d = 0; d < 4; ++d) {
        v16h bk = *(const v16h*)&sK[16 + ll][d * 32 + hl * 16];
        sc1 = wmma_f16(aq[d], bk, sc1);
      }
      // ---- mask + online softmax + P to LDS (A-layout re-shuffle) ----
#pragma unroll
      for (int r = 0; r < 8; ++r) {
        int row = q0w + hl * 8 + r;
        int c0 = k0 + ll, c1 = k0 + 16 + ll;
        float s0 = (c0 <= row) ? sc0[r] * scale : -1e30f;
        float s1 = (c1 <= row) ? sc1[r] * scale : -1e30f;
        float mx = fmaxf(s0, s1);
#pragma unroll
        for (int off = 8; off >= 1; off >>= 1)
          mx = fmaxf(mx, __shfl_xor(mx, off, 32));
        float mnew = fmaxf(m[r], mx);
        float f = __expf(m[r] - mnew);
        m[r] = mnew;
        float p0 = __expf(s0 - mnew);
        float p1 = __expf(s1 - mnew);
        float ps = p0 + p1;
#pragma unroll
        for (int off = 8; off >= 1; off >>= 1)
          ps += __shfl_xor(ps, off, 32);
        l[r] = l[r] * f + ps;
#pragma unroll
        for (int dt = 0; dt < 8; ++dt) oacc[dt][r] *= f;
        sP[wave][hl * 8 + r][ll]      = (_Float16)p0;
        sP[wave][hl * 8 + r][16 + ll] = (_Float16)p1;
      }
    }
    __syncthreads();
    if (active) {
      // ---- PV: A = P (16q x 32key), B = V (32key x 16d) from sV ----
      v16h ap = *(const v16h*)&sP[wave][ll][hl * 16];
#pragma unroll
      for (int dt = 0; dt < 8; ++dt) {
        v16h bv = *(const v16h*)&sV[dt * 16 + ll][hl * 16];
        oacc[dt] = wmma_f16(ap, bv, oacc[dt]);
      }
    }
  }
  // ---- normalize + store attn f16 [s, h*128+d] ----
#pragma unroll
  for (int r = 0; r < 8; ++r) {
    float inv_l = 1.0f / l[r];
    int row = q0w + hl * 8 + r;
#pragma unroll
    for (int dt = 0; dt < 8; ++dt)
      attn[(size_t)row * HID + h * HDIM + dt * 16 + ll] =
          (_Float16)(oacc[dt][r] * inv_l);
  }
}

// ---------------------------------------------------------------------------
extern "C" void kernel_launch(void* const* d_in, const int* in_sizes, int n_in,
                              void* d_out, int out_size, void* d_ws, size_t ws_size,
                              hipStream_t stream) {
  const float* hidden = (const float*)d_in[0];
  const int*   pos    = (const int*)d_in[1];
  const float* Wq     = (const float*)d_in[2];
  const float* Wk     = (const float*)d_in[3];
  const float* Wv     = (const float*)d_in[4];
  const float* Wo     = (const float*)d_in[5];

  char* ws = (char*)d_ws;
  _Float16* qf   = (_Float16*)(ws);                           // 16 MiB
  _Float16* kf   = (_Float16*)(ws + (size_t)16777216);        //  4 MiB
  _Float16* vf   = (_Float16*)(ws + (size_t)20971520);        //  4 MiB
  _Float16* attn = (_Float16*)(ws + (size_t)25165824);        // 16 MiB

  // Projections (fp32 A -> f16 out through WMMA)
  gemm_tile<false, true><<<dim3(S_LEN / 64, HID / 64),   256, 0, stream>>>(hidden, Wq, qf, S_LEN, HID,   HID);
  gemm_tile<false, true><<<dim3(S_LEN / 64, KVHID / 64), 256, 0, stream>>>(hidden, Wk, kf, S_LEN, KVHID, HID);
  gemm_tile<false, true><<<dim3(S_LEN / 64, KVHID / 64), 256, 0, stream>>>(hidden, Wv, vf, S_LEN, KVHID, HID);

  // RoPE in place on q and k
  rope_kernel<<<(S_LEN * NHEAD * 64) / 256, 256, 0, stream>>>(qf, pos, NHEAD, S_LEN * NHEAD * 64);
  rope_kernel<<<(S_LEN * NKV * 64) / 256,   256, 0, stream>>>(kf, pos, NKV,   S_LEN * NKV * 64);

  // Causal flash attention
  flash_attn<<<dim3(S_LEN / 128, NHEAD), 256, 0, stream>>>(qf, kf, vf, attn);

  // Output projection (f16 A -> fp32 out)
  gemm_tile<true, false><<<dim3(S_LEN / 64, HID / 64), 256, 0, stream>>>(attn, Wo, d_out, S_LEN, HID, HID);
}
